// TransformerBlock_3753801417193
// MI455X (gfx1250) — matrix operand order repaired and checked by reading, not compiled, not run on hardware
//
#include <hip/hip_runtime.h>
#include <hip/hip_bf16.h>
#include <cstdint>

#define B_  8
#define S_  1024
#define D_  1024
#define H_  16
#define HD_ 64
#define DF_ 4096
#define M_  (B_ * S_)   // 8192 rows (tokens)

typedef __attribute__((ext_vector_type(16))) __bf16 v16bf;
typedef __attribute__((ext_vector_type(8)))  float  v8f;
typedef unsigned short u16;

union FragBF {
    v16bf bf;
    int4  i4[2];
    u16   us[16];
};

__device__ __forceinline__ u16 f2bf(float f) {
    unsigned u = __float_as_uint(f);
    u += 0x7FFFu + ((u >> 16) & 1u);          // round to nearest even
    return (u16)(u >> 16);
}

// A-matrix 16x32 bf16 fragment (row-major A, lane = M row).
// lanes 0-15: K = 0..7 in v0-3, 16..23 in v4-7 ; lanes 16-31: K = 8..15, 24..31
__device__ __forceinline__ v16bf load_fragA(const u16* base, int ld, int kb, int lane) {
    int row = lane & 15;
    int ks  = (lane < 16) ? 0 : 8;
    const u16* p = base + (size_t)row * ld + kb + ks;
    FragBF f;
    f.i4[0] = *(const int4*)(p);
    f.i4[1] = *(const int4*)(p + 16);
    return f.bf;
}

// B-matrix 32x16 bf16 fragment, sourced from row-major W[n, k] (i.e. B^T).
// lane = N col; lanes 0-15 hold K=0..7,16..23, lanes 16-31 hold K=8..15,24..31
__device__ __forceinline__ v16bf load_fragB(const u16* base, int ld, int kb, int lane) {
    int row = lane & 15;
    int ks  = (lane < 16) ? 0 : 8;
    const u16* p = base + (size_t)row * ld + kb + ks;
    FragBF f;
    f.i4[0] = *(const int4*)(p);
    f.i4[1] = *(const int4*)(p + 16);
    return f.bf;
}

__device__ __forceinline__ v8f wmma_bf16(v16bf a, v16bf b, v8f c) {
    return __builtin_amdgcn_wmma_f32_16x16x32_bf16(false, a, false, b, (short)0, c, false, false);
}

// Async 16-byte global -> LDS copy (ASYNCcnt-tracked). The LDS operand is the
// low 32 bits of the generic pointer (flat LDS aperture keeps the DS offset in
// addr[31:0]).
__device__ __forceinline__ void async_cp16(void* lds, const void* gptr) {
    unsigned loff = (unsigned)(uintptr_t)lds;
    asm volatile("global_load_async_to_lds_b128 %0, %1, off"
                 :: "v"(loff), "v"(gptr) : "memory");
}

template <int N>
__device__ __forceinline__ void wait_async() {
    asm volatile("s_wait_asynccnt %0" :: "i"(N) : "memory");
}

// ---------------------------------------------------------------------------
// LDS-staged, double-buffered GEMM:
//   C[M,N] = A[M,K] (bf16) @ W[N,K]^T (bf16) + bias + epilogue.
// Block = 256 threads (8 waves) -> macro tile 64(M) x 128(N), K-step 64.
// Stage i+1 is streamed into the alternate LDS buffer with
// global_load_async_to_lds_b128 while stage i feeds 8 WMMAs per wave;
// s_wait_asynccnt 6 exploits in-order async completion to wait only for the
// previous stage's 6 copies.
// EPI 0: outB = bf16((acc + bias) * scale)
// EPI 1: outF = acc + bias + resid           (fp32 out)
// EPI 2: outB = bf16(gelu_exact(acc + bias))
// ---------------------------------------------------------------------------
template <int EPI>
__global__ __launch_bounds__(256) void gemm_kernel(
    const u16* __restrict__ A, const u16* __restrict__ W,
    const float* __restrict__ bias, const float* __restrict__ resid,
    float* __restrict__ outF, u16* __restrict__ outB,
    int N, int K, float scale)
{
    constexpr int BM = 64, BN = 128, BK = 64;
    __shared__ u16 smA[2][BM * BK];   // 2 x  8 KB
    __shared__ u16 smB[2][BN * BK];   // 2 x 16 KB

    int tid  = threadIdx.x;
    int wave = tid >> 5;
    int lane = tid & 31;
    int m0 = blockIdx.y * BM;
    int n0 = blockIdx.x * BN;
    int wm = (wave >> 2) * 32;     // wave M offset inside tile (0 / 32)
    int wn = (wave & 3)  * 32;     // wave N offset inside tile (0..96)

    // Staging assignment: 8-u16 (16B) segments. A: 512 segs, B: 1024 segs.
    const u16* agp[2];  int aofs[2];
    const u16* bgp[4];  int bofs[4];
#pragma unroll
    for (int i = 0; i < 2; ++i) {
        int s = tid + i * 256;
        int r = s >> 3, c = (s & 7) * 8;
        agp[i]  = A + (size_t)(m0 + r) * K + c;
        aofs[i] = r * BK + c;
    }
#pragma unroll
    for (int i = 0; i < 4; ++i) {
        int s = tid + i * 256;
        int r = s >> 3, c = (s & 7) * 8;
        bgp[i]  = W + (size_t)(n0 + r) * K + c;
        bofs[i] = r * BK + c;
    }

    // Prologue: stage 0 into buffer 0.
#pragma unroll
    for (int i = 0; i < 2; ++i) async_cp16(&smA[0][aofs[i]], agp[i]);
#pragma unroll
    for (int i = 0; i < 4; ++i) async_cp16(&smB[0][bofs[i]], bgp[i]);

    v8f acc00 = {}, acc01 = {}, acc10 = {}, acc11 = {};

    int buf = 0;
    for (int kb = 0; kb < K; kb += BK, buf ^= 1) {
        int kn = kb + BK;
        if (kn < K) {
            // Stream next stage into the alternate buffer (6 async copies),
            // then wait for the previous 6 (in-order completion).
#pragma unroll
            for (int i = 0; i < 2; ++i) async_cp16(&smA[buf ^ 1][aofs[i]], agp[i] + kn);
#pragma unroll
            for (int i = 0; i < 4; ++i) async_cp16(&smB[buf ^ 1][bofs[i]], bgp[i] + kn);
            wait_async<6>();
        } else {
            wait_async<0>();
        }
        __syncthreads();

        // Consume: 2x2 C-tiles x 2 K-fragments = 8 WMMA.
#pragma unroll
        for (int kk = 0; kk < BK; kk += 32) {
            v16bf a0 = load_fragA(&smA[buf][(wm +  0) * BK], BK, kk, lane);
            v16bf a1 = load_fragA(&smA[buf][(wm + 16) * BK], BK, kk, lane);
            v16bf b0 = load_fragB(&smB[buf][(wn +  0) * BK], BK, kk, lane);
            v16bf b1 = load_fragB(&smB[buf][(wn + 16) * BK], BK, kk, lane);
            acc00 = wmma_bf16(a0, b0, acc00);
            acc01 = wmma_bf16(a0, b1, acc01);
            acc10 = wmma_bf16(a1, b0, acc10);
            acc11 = wmma_bf16(a1, b1, acc11);
        }
        __syncthreads();   // all waves done reading buf before it is re-staged
    }

    int rowoff = (lane < 16) ? 0 : 8;
    v8f accs[2][2] = {{acc00, acc01}, {acc10, acc11}};
#pragma unroll
    for (int ar = 0; ar < 2; ++ar) {
#pragma unroll
        for (int br = 0; br < 2; ++br) {
            int n = n0 + wn + br * 16 + (lane & 15);
            float bv = bias[n];
#pragma unroll
            for (int r = 0; r < 8; ++r) {
                int m = m0 + wm + ar * 16 + rowoff + r;
                size_t idx = (size_t)m * N + n;
                float v = accs[ar][br][r] + bv;
                if (EPI == 0) {
                    outB[idx] = f2bf(v * scale);
                } else if (EPI == 1) {
                    outF[idx] = v + resid[idx];
                } else {
                    float g = 0.5f * v * (1.0f + erff(v * 0.70710678118f));
                    outB[idx] = f2bf(g);
                }
            }
        }
    }
}

// ---------------------------------------------------------------------------
// LayerNorm over D=1024; one block (256 thr) per row. Writes fp32 (optional)
// and bf16 copies.
// ---------------------------------------------------------------------------
__global__ __launch_bounds__(256) void ln_kernel(
    const float* __restrict__ x, const float* __restrict__ g,
    const float* __restrict__ bta, float* __restrict__ outF, u16* __restrict__ outB)
{
    __shared__ float red[256];
    int row = blockIdx.x;
    int tid = threadIdx.x;
    const float* xr = x + (size_t)row * D_;

    float v[4];
    float s = 0.f;
#pragma unroll
    for (int i = 0; i < 4; ++i) { v[i] = xr[tid + i * 256]; s += v[i]; }
    red[tid] = s; __syncthreads();
    for (int o = 128; o > 0; o >>= 1) { if (tid < o) red[tid] += red[tid + o]; __syncthreads(); }
    float mu = red[0] * (1.0f / D_);
    __syncthreads();

    float q = 0.f;
#pragma unroll
    for (int i = 0; i < 4; ++i) { float d = v[i] - mu; q += d * d; }
    red[tid] = q; __syncthreads();
    for (int o = 128; o > 0; o >>= 1) { if (tid < o) red[tid] += red[tid + o]; __syncthreads(); }
    float rs = rsqrtf(red[0] * (1.0f / D_) + 1e-5f);

#pragma unroll
    for (int i = 0; i < 4; ++i) {
        int c = tid + i * 256;
        float o = (v[i] - mu) * rs * g[c] + bta[c];
        if (outF) outF[(size_t)row * D_ + c] = o;
        outB[(size_t)row * D_ + c] = f2bf(o);
    }
}

__global__ void cast_kernel(const float* __restrict__ in, u16* __restrict__ out, int n) {
    int i = blockIdx.x * 256 + threadIdx.x;
    if (i < n) out[i] = f2bf(in[i]);
}

__global__ void zero_kernel(float* __restrict__ p, int n) {
    int i = blockIdx.x * 256 + threadIdx.x;
    if (i < n) p[i] = 0.f;
}

// vT[b,h,d,s] = v[b,s,h*HD+d]
__global__ void vtrans_kernel(const u16* __restrict__ v, u16* __restrict__ vT) {
    int idx = blockIdx.x * 256 + threadIdx.x;      // B*H*HD*S = 8M
    int s2 = idx & (S_ - 1);
    int d  = (idx >> 10) & (HD_ - 1);
    int h  = (idx >> 16) & (H_ - 1);
    int b  = idx >> 20;
    vT[idx] = v[((size_t)b * S_ + s2) * D_ + h * HD_ + d];
}

// ---------------------------------------------------------------------------
// Attention: block = 128 threads (4 waves), one (b, h, 16-row q tile).
// Phase 1: scores 16x1024 via WMMA into LDS (64KB dynamic).
// Phase 2: softmax fp32 + attn_mat atomic accumulation (mean over heads).
// Phase 3: ctx = probs @ V via WMMA against transposed V.
// ---------------------------------------------------------------------------
__global__ __launch_bounds__(128) void attn_kernel(
    const u16* __restrict__ q, const u16* __restrict__ k,
    const u16* __restrict__ vT, float* __restrict__ attn_mat,
    u16* __restrict__ ctx)
{
    extern __shared__ float sc[];                  // 16 x 1024 fp32 scores/probs
    int tid  = threadIdx.x;
    int wave = tid >> 5;
    int lane = tid & 31;
    int q0 = blockIdx.x * 16;
    int h  = blockIdx.y;
    int b  = blockIdx.z;

    // ---- Phase 1: scores = q_scaled @ k^T ----
    const u16* qbase = q + ((size_t)b * S_ + q0) * D_ + h * HD_;
    v16bf a0 = load_fragA(qbase, D_, 0,  lane);
    v16bf a1 = load_fragA(qbase, D_, 32, lane);
    int rowoff = (lane < 16) ? 0 : 8;

    for (int nt = 0; nt < 16; ++nt) {
        int n0 = wave * 256 + nt * 16;
        const u16* kbase = k + ((size_t)b * S_ + n0) * D_ + h * HD_;
        v8f acc = {};
        acc = wmma_bf16(a0, load_fragB(kbase, D_, 0,  lane), acc);
        acc = wmma_bf16(a1, load_fragB(kbase, D_, 32, lane), acc);
        int n = n0 + (lane & 15);
#pragma unroll
        for (int r = 0; r < 8; ++r) sc[(rowoff + r) * S_ + n] = acc[r];
    }
    __syncthreads();

    // ---- Phase 2: softmax per row (8 threads per row), attn_mat accum ----
    {
        int row = tid >> 3;
        int g   = tid & 7;
        float* srow = sc + row * S_;
        float mx = -3.0e38f;
        for (int c = g; c < S_; c += 8) mx = fmaxf(mx, srow[c]);
        for (int o = 1; o < 8; o <<= 1) mx = fmaxf(mx, __shfl_xor(mx, o, 8));
        float sum = 0.f;
        for (int c = g; c < S_; c += 8) { float e = __expf(srow[c] - mx); srow[c] = e; sum += e; }
        for (int o = 1; o < 8; o <<= 1) sum += __shfl_xor(sum, o, 8);
        float inv = 1.0f / sum;
        float* am = attn_mat + ((size_t)b * S_ + q0 + row) * S_;
        for (int c = g; c < S_; c += 8) {
            float p = srow[c] * inv;
            srow[c] = p;
            atomicAdd(&am[c], p * (1.0f / H_));
        }
    }
    __syncthreads();

    // ---- Phase 3: ctx = probs @ V  (wave handles 16 of the 64 head dims) ----
    int dw = wave * 16;
    const u16* vbase = vT + (((size_t)b * H_ + h) * HD_ + dw) * S_;
    v8f acc = {};
    int arow = lane & 15;
    int ks   = (lane < 16) ? 0 : 8;
    for (int kb = 0; kb < S_; kb += 32) {
        const float* sp = sc + arow * S_ + kb + ks;
        FragBF af;
#pragma unroll
        for (int j = 0; j < 8; ++j) {
            af.us[j]     = f2bf(sp[j]);
            af.us[8 + j] = f2bf(sp[16 + j]);
        }
        v16bf bf = load_fragB(vbase, S_, kb, lane);
        acc = wmma_bf16(af.bf, bf, acc);
    }
    int n = dw + (lane & 15);
#pragma unroll
    for (int r = 0; r < 8; ++r) {
        ctx[((size_t)b * S_ + q0 + rowoff + r) * D_ + h * HD_ + n] = f2bf(acc[r]);
    }
}

// ---------------------------------------------------------------------------
extern "C" void kernel_launch(void* const* d_in, const int* in_sizes, int n_in,
                              void* d_out, int out_size, void* d_ws, size_t ws_size,
                              hipStream_t stream) {
    (void)in_sizes; (void)n_in; (void)out_size; (void)ws_size;
    const float* query     = (const float*)d_in[0];
    const float* in_proj_w = (const float*)d_in[1];
    const float* in_proj_b = (const float*)d_in[2];
    const float* out_w     = (const float*)d_in[3];
    const float* out_b     = (const float*)d_in[4];
    const float* ln1_g     = (const float*)d_in[5];
    const float* ln1_b     = (const float*)d_in[6];
    const float* ln2_g     = (const float*)d_in[7];
    const float* ln2_b     = (const float*)d_in[8];
    const float* fc1_w     = (const float*)d_in[9];
    const float* fc1_b     = (const float*)d_in[10];
    const float* fc2_w     = (const float*)d_in[11];
    const float* fc2_b     = (const float*)d_in[12];

    float* x_out    = (float*)d_out;                       // B*S*D
    float* attn_mat = x_out + (size_t)B_ * S_ * D_;        // B*S*S

    char* ws = (char*)d_ws;
    size_t off = 0;
    auto alloc = [&](size_t bytes) -> void* {
        void* p = ws + off;
        off = (off + bytes + 255) & ~(size_t)255;
        return p;
    };

    float* qn_f = (float*)alloc((size_t)M_ * D_ * 4);
    float* x_f  = (float*)alloc((size_t)M_ * D_ * 4);
    u16* qn_b   = (u16*)alloc((size_t)M_ * D_ * 2);
    u16* kv_b   = (u16*)alloc((size_t)M_ * D_ * 2);
    u16* q_b    = (u16*)alloc((size_t)M_ * D_ * 2);
    u16* k_b    = (u16*)alloc((size_t)M_ * D_ * 2);
    u16* v_b    = (u16*)alloc((size_t)M_ * D_ * 2);
    u16* vT_b   = (u16*)alloc((size_t)M_ * D_ * 2);
    u16* ctx_b  = (u16*)alloc((size_t)M_ * D_ * 2);
    u16* h_b    = (u16*)alloc((size_t)M_ * D_ * 2);
    u16* h1_b   = (u16*)alloc((size_t)M_ * DF_ * 2);
    u16* w_in_b  = (u16*)alloc((size_t)3 * D_ * D_ * 2);
    u16* w_out_b = (u16*)alloc((size_t)D_ * D_ * 2);
    u16* w_fc1_b = (u16*)alloc((size_t)DF_ * D_ * 2);
    u16* w_fc2_b = (u16*)alloc((size_t)D_ * DF_ * 2);

    // Zero attn_mat (it is accumulated with atomics).
    zero_kernel<<<(B_ * S_ * S_) / 256, 256, 0, stream>>>(attn_mat, B_ * S_ * S_);

    // Cast activations + weights to bf16.
    cast_kernel<<<(M_ * D_) / 256, 256, 0, stream>>>(query, kv_b, M_ * D_);
    cast_kernel<<<(3 * D_ * D_) / 256, 256, 0, stream>>>(in_proj_w, w_in_b, 3 * D_ * D_);
    cast_kernel<<<(D_ * D_) / 256, 256, 0, stream>>>(out_w, w_out_b, D_ * D_);
    cast_kernel<<<(DF_ * D_) / 256, 256, 0, stream>>>(fc1_w, w_fc1_b, DF_ * D_);
    cast_kernel<<<(D_ * DF_) / 256, 256, 0, stream>>>(fc2_w, w_fc2_b, D_ * DF_);

    // LN1 -> qn (fp32 + bf16)
    ln_kernel<<<M_, 256, 0, stream>>>(query, ln1_g, ln1_b, qn_f, qn_b);

    // QKV projections. q gets 1/sqrt(HD) folded in.
    dim3 gD(D_ / 128, M_ / 64);
    gemm_kernel<0><<<gD, 256, 0, stream>>>(qn_b, w_in_b, in_proj_b, nullptr,
                                           nullptr, q_b, D_, D_, 0.125f);
    gemm_kernel<0><<<gD, 256, 0, stream>>>(kv_b, w_in_b + (size_t)D_ * D_, in_proj_b + D_,
                                           nullptr, nullptr, k_b, D_, D_, 1.0f);
    gemm_kernel<0><<<gD, 256, 0, stream>>>(kv_b, w_in_b + (size_t)2 * D_ * D_, in_proj_b + 2 * D_,
                                           nullptr, nullptr, v_b, D_, D_, 1.0f);

    // Transpose V for context GEMM B-operand.
    vtrans_kernel<<<(M_ * D_) / 256, 256, 0, stream>>>(v_b, vT_b);

    // Attention (scores -> softmax -> attn_mat -> ctx).
    dim3 gA(S_ / 16, H_, B_);
    attn_kernel<<<gA, 128, 16 * S_ * sizeof(float), stream>>>(q_b, k_b, vT_b, attn_mat, ctx_b);

    // Out projection + residual (x = qn + attn_out).
    gemm_kernel<1><<<gD, 256, 0, stream>>>(ctx_b, w_out_b, out_b, qn_f,
                                           x_f, nullptr, D_, D_, 1.0f);

    // LN2 -> h (bf16 only)
    ln_kernel<<<M_, 256, 0, stream>>>(x_f, ln2_g, ln2_b, nullptr, h_b);

    // FC1 + exact GELU.
    dim3 gF1(DF_ / 128, M_ / 64);
    gemm_kernel<2><<<gF1, 256, 0, stream>>>(h_b, w_fc1_b, fc1_b, nullptr,
                                            nullptr, h1_b, DF_, D_, 1.0f);

    // FC2 + residual -> final x into d_out.
    gemm_kernel<1><<<gD, 256, 0, stream>>>(h1_b, w_fc2_b, fc2_b, x_f,
                                           x_out, nullptr, D_, DF_, 1.0f);
}
